// LINKX_23596550324872
// MI455X (gfx1250) — matrix operand order repaired and checked by reading, not compiled, not run on hardware
//
#include <hip/hip_runtime.h>
#include <hip/hip_bf16.h>

// ---------------------------------------------------------------------------
// LINKX forward on MI455X (gfx1250, wave32).
// GEMMs via v_wmma_f32_16x16x32_bf16; one wave owns a 16-row strip and all
// column tiles (A loaded once per K-slab, 8 WMMAs issued back-to-back).
// Weights pre-converted to bf16 so B fragments are single 32B loads.
// Sparse aggregation via coalesced b128 gathers + global f32 atomics.
// ---------------------------------------------------------------------------

typedef __attribute__((ext_vector_type(16))) __bf16 v16bf;
typedef __attribute__((ext_vector_type(4)))  __bf16 v4bf;
typedef __attribute__((ext_vector_type(8)))  float  v8f;

#define H_DIM 128
#define BN_EPS 1e-5f

// ---------------------------------------------------------------------------
__global__ void k_init_misc(int* rmin, float* sum, float* sumsq) {
    int t = threadIdx.x;
    if (t == 0) *rmin = 0x7FFFFFFF;
    if (t < H_DIM) { sum[t] = 0.0f; sumsq[t] = 0.0f; }
}

// xA half of interleaved [N,256] buffer preset to bias bA (float4 stores)
__global__ void k_init_xA(float* __restrict__ xAX, const float* __restrict__ bA, int N) {
    int idx = blockIdx.x * blockDim.x + threadIdx.x;     // N * 32 threads
    if (idx >= N * 32) return;
    int i = idx >> 5, h = (idx & 31) * 4;
    float4 b = *reinterpret_cast<const float4*>(bA + h);
    *reinterpret_cast<float4*>(xAX + (size_t)i * 256 + h) = b;
}

__global__ void k_row_min(const int* __restrict__ row, int E, int* rmin) {
    int m = 0x7FFFFFFF;
    for (long long i = (long long)blockIdx.x * blockDim.x + threadIdx.x; i < E;
         i += (long long)gridDim.x * blockDim.x)
        m = min(m, row[i]);
    atomicMin(rmin, m);
}

// weight f32 -> bf16 (4 elements / thread, b128 in, b64 out)
__global__ void k_cvt_bf16(const float* __restrict__ src, __bf16* __restrict__ dst, int n4) {
    int i = blockIdx.x * blockDim.x + threadIdx.x;
    if (i >= n4) return;
    float4 v = reinterpret_cast<const float4*>(src)[i];
    v4bf o = { (__bf16)v.x, (__bf16)v.y, (__bf16)v.z, (__bf16)v.w };
    reinterpret_cast<v4bf*>(dst)[i] = o;
}

// WAt[n][h] = WA[h][n]   (WA: [128, N] row-major) — LDS tiled transpose
__global__ void k_transpose_WA(const float* __restrict__ WA, float* __restrict__ WAt, int N) {
    __shared__ float tile[32][33];
    int n0 = blockIdx.x * 32;
    int h0 = blockIdx.y * 32;
    int tx = threadIdx.x;
    for (int i = threadIdx.y; i < 32; i += 8) {
        int h = h0 + i, n = n0 + tx;
        tile[i][tx] = (n < N) ? WA[(size_t)h * N + n] : 0.0f;
    }
    __syncthreads();
    for (int i = threadIdx.y; i < 32; i += 8) {
        int n = n0 + i, h = h0 + tx;
        if (n < N) WAt[(size_t)n * H_DIM + h] = tile[tx][i];
    }
}

// xAX[row-rmin][4q..4q+3] += WAt[col][4q..4q+3]  — wave-uniform edge index,
// b128 gather + 4 f32 atomics per thread (32 threads per edge)
__global__ void k_scatter_xA(const int* __restrict__ edge, int E,
                             const float* __restrict__ WAt,
                             float* __restrict__ xAX,
                             const int* __restrict__ rmin) {
    long long idx = (long long)blockIdx.x * blockDim.x + threadIdx.x;
    if (idx >= (long long)E * 32) return;
    int e = (int)(idx >> 5);
    int h = (int)(idx & 31) * 4;
    int r = edge[e] - *rmin;             // edge_index[0][e] - row.min()
    int c = edge[E + e];                 // edge_index[1][e]
    float4 w = *reinterpret_cast<const float4*>(WAt + (size_t)c * H_DIM + h);
    float* dst = xAX + (size_t)r * 256 + h;
    atomicAdd(dst + 0, w.x);
    atomicAdd(dst + 1, w.y);
    atomicAdd(dst + 2, w.z);
    atomicAdd(dst + 3, w.w);
}

// ---------------------------------------------------------------------------
// WMMA GEMM:  C[:, c_off + 0..NT*16) = epi( A[MxK] @ Wb[NT*16 x K]^T + bias )
// 4 waves / block; each wave owns one 16-row strip and ALL NT column tiles.
// A fragment loaded once per 32-wide K slab (4x b128), reused for NT WMMAs.
// EPI: 0 = bias, 1 = relu, 2 = relu(bias + xA + xX residual)
// BN_A: per-column scale/shift on A before bf16 conversion (fused batchnorm).
template <int NT, int EPI, bool BN_A>
__global__ __launch_bounds__(128) void k_wmma_gemm(
        const float*  __restrict__ A, int lda,
        const __bf16* __restrict__ Wb, int K,
        const float*  __restrict__ bias,
        const float*  __restrict__ addsrc, int ldadd,
        const float*  __restrict__ scale,
        const float*  __restrict__ shift,
        float* __restrict__ C, int ldc, int c_off, int M) {
    const int wave = threadIdx.x >> 5;
    const int lane = threadIdx.x & 31;
    const int m0   = (blockIdx.x * 4 + wave) * 16;
    if (m0 >= M) return;                 // wave-uniform: EXEC stays all-ones

    const int mrow = lane & 15;
    const int hi   = lane >> 4;

    v8f acc[NT];
#pragma unroll
    for (int t = 0; t < NT; ++t) acc[t] = (v8f){0,0,0,0,0,0,0,0};

    const float* __restrict__ Abase = A + (size_t)(m0 + mrow) * lda;

    for (int k0 = 0; k0 < K; k0 += 32) {
        if (k0 + 32 < K) {               // global_prefetch_b8 next K slab
            __builtin_prefetch(Abase + k0 + 32, 0, 3);
            __builtin_prefetch(Abase + k0 + 56, 0, 3);
        }
        // ---- A fragment: one contiguous 16-float run per lane ----
        // run covers kk = 16*hi + [0,16)
        float4 a0 = *reinterpret_cast<const float4*>(Abase + k0 + 16 * hi);
        float4 a1 = *reinterpret_cast<const float4*>(Abase + k0 + 16 * hi + 4);
        float4 a2 = *reinterpret_cast<const float4*>(Abase + k0 + 16 * hi + 8);
        float4 a3 = *reinterpret_cast<const float4*>(Abase + k0 + 16 * hi + 12);
        if (BN_A) {                      // fused training-mode batchnorm on A
            const float4 s0 = *reinterpret_cast<const float4*>(scale + k0 + 16 * hi);
            const float4 s1 = *reinterpret_cast<const float4*>(scale + k0 + 16 * hi + 4);
            const float4 s2 = *reinterpret_cast<const float4*>(scale + k0 + 16 * hi + 8);
            const float4 s3 = *reinterpret_cast<const float4*>(scale + k0 + 16 * hi + 12);
            const float4 t0 = *reinterpret_cast<const float4*>(shift + k0 + 16 * hi);
            const float4 t1 = *reinterpret_cast<const float4*>(shift + k0 + 16 * hi + 4);
            const float4 t2 = *reinterpret_cast<const float4*>(shift + k0 + 16 * hi + 8);
            const float4 t3 = *reinterpret_cast<const float4*>(shift + k0 + 16 * hi + 12);
            a0.x = fmaf(a0.x, s0.x, t0.x); a0.y = fmaf(a0.y, s0.y, t0.y);
            a0.z = fmaf(a0.z, s0.z, t0.z); a0.w = fmaf(a0.w, s0.w, t0.w);
            a1.x = fmaf(a1.x, s1.x, t1.x); a1.y = fmaf(a1.y, s1.y, t1.y);
            a1.z = fmaf(a1.z, s1.z, t1.z); a1.w = fmaf(a1.w, s1.w, t1.w);
            a2.x = fmaf(a2.x, s2.x, t2.x); a2.y = fmaf(a2.y, s2.y, t2.y);
            a2.z = fmaf(a2.z, s2.z, t2.z); a2.w = fmaf(a2.w, s2.w, t2.w);
            a3.x = fmaf(a3.x, s3.x, t3.x); a3.y = fmaf(a3.y, s3.y, t3.y);
            a3.z = fmaf(a3.z, s3.z, t3.z); a3.w = fmaf(a3.w, s3.w, t3.w);
        }
        v16bf a;
        a[0]  = (__bf16)a0.x; a[1]  = (__bf16)a0.y; a[2]  = (__bf16)a0.z; a[3]  = (__bf16)a0.w;
        a[4]  = (__bf16)a1.x; a[5]  = (__bf16)a1.y; a[6]  = (__bf16)a1.z; a[7]  = (__bf16)a1.w;
        a[8]  = (__bf16)a2.x; a[9]  = (__bf16)a2.y; a[10] = (__bf16)a2.z; a[11] = (__bf16)a2.w;
        a[12] = (__bf16)a3.x; a[13] = (__bf16)a3.y; a[14] = (__bf16)a3.z; a[15] = (__bf16)a3.w;

        // ---- NT back-to-back WMMAs, B frag = one aligned 32B bf16 load ----
#pragma unroll
        for (int t = 0; t < NT; ++t) {
            const v16bf b = *reinterpret_cast<const v16bf*>(
                Wb + (size_t)(t * 16 + mrow) * K + k0 + 16 * hi);
            acc[t] = __builtin_amdgcn_wmma_f32_16x16x32_bf16(
                false, a, false, b, (short)0, acc[t], false, false);
        }
    }

    // ---- epilogue (C layout: VGPR v -> row v + 8*hi, col lane&15) ----
    const int nn = lane & 15;
#pragma unroll
    for (int t = 0; t < NT; ++t) {
        const int n  = t * 16 + nn;
        const float bv = bias[n];
#pragma unroll
        for (int v = 0; v < 8; ++v) {
            int m = m0 + v + 8 * hi;
            float val = acc[t][v] + bv;
            if (EPI == 2) {
                val += addsrc[(size_t)m * ldadd + n] +
                       addsrc[(size_t)m * ldadd + H_DIM + n];
                val = fmaxf(val, 0.0f);
            } else if (EPI == 1) {
                val = fmaxf(val, 0.0f);
            }
            C[(size_t)m * ldc + c_off + n] = val;
        }
    }
}

// ---------------------------------------------------------------------------
__global__ void k_bn_stats(const float* __restrict__ h, float* sum, float* sumsq, int M) {
    int j   = threadIdx.x;               // 128 threads = 128 features
    int rpb = (M + gridDim.x - 1) / gridDim.x;
    int r0  = blockIdx.x * rpb;
    int r1  = min(M, r0 + rpb);
    float s = 0.0f, s2 = 0.0f;
    for (int r = r0; r < r1; ++r) {
        float v = h[(size_t)r * H_DIM + j];
        s += v; s2 += v * v;
    }
    atomicAdd(&sum[j], s);
    atomicAdd(&sumsq[j], s2);
}

__global__ void k_bn_coeffs(const float* sum, const float* sumsq,
                            const float* gamma, const float* beta,
                            float* scale, float* shift, int M) {
    int j = threadIdx.x;
    float inv  = 1.0f / (float)M;
    float mean = sum[j] * inv;
    float var  = sumsq[j] * inv - mean * mean;
    float sc   = gamma[j] * rsqrtf(var + BN_EPS);
    scale[j] = sc;
    shift[j] = beta[j] - mean * sc;
}

// ---------------------------------------------------------------------------
extern "C" void kernel_launch(void* const* d_in, const int* in_sizes, int n_in,
                              void* d_out, int out_size, void* d_ws, size_t ws_size,
                              hipStream_t stream) {
    const float* x    = (const float*)d_in[0];
    const int*   edge = (const int*)  d_in[1];
    const float* WA   = (const float*)d_in[2];
    const float* bA   = (const float*)d_in[3];
    const float* WX   = (const float*)d_in[4];
    const float* bX   = (const float*)d_in[5];
    const float* W    = (const float*)d_in[6];
    const float* bW   = (const float*)d_in[7];
    const float* Wf1  = (const float*)d_in[8];
    const float* bf1  = (const float*)d_in[9];
    const float* gamma= (const float*)d_in[10];
    const float* beta = (const float*)d_in[11];
    const float* Wf2  = (const float*)d_in[12];
    const float* bf2  = (const float*)d_in[13];
    float* out = (float*)d_out;

    const int N  = 50000;                // 3125 strips of 16 rows
    const int IN = 512;
    const int E  = in_sizes[1] / 2;      // edge_index is (2, E)

    // workspace carve-up
    char*   ws    = (char*)d_ws;
    int*    rmin  = (int*)  ws;
    float*  sum   = (float*)(ws + 256);
    float*  sumsq = (float*)(ws + 768);
    float*  scale = (float*)(ws + 1280);
    float*  shift = (float*)(ws + 1792);
    float*  WAt   = (float*)(ws + 4096);           // [N, 128]
    float*  xAX   = WAt  + (size_t)N * H_DIM;      // [N, 256] interleaved xA|xX
    float*  hbuf  = xAX  + (size_t)N * 256;        // [N, 128]
    float*  h2    = hbuf + (size_t)N * H_DIM;      // [N, 128]
    __bf16* WXb   = (__bf16*)(h2 + (size_t)N * H_DIM); // [128, 512] bf16
    __bf16* Wb    = WXb  + (size_t)H_DIM * IN;     // [128, 256] bf16
    __bf16* Wf1b  = Wb   + (size_t)H_DIM * 256;    // [128, 128] bf16
    __bf16* Wf2b  = Wf1b + (size_t)H_DIM * H_DIM;  // [ 64, 128] bf16

    // 1) init + weight conversion
    k_init_misc<<<1, 128, 0, stream>>>(rmin, sum, sumsq);
    k_init_xA<<<(N * 32 + 255) / 256, 256, 0, stream>>>(xAX, bA, N);
    k_cvt_bf16<<<(H_DIM * IN / 4 + 255) / 256,    256, 0, stream>>>(WX,  WXb,  H_DIM * IN / 4);
    k_cvt_bf16<<<(H_DIM * 256 / 4 + 255) / 256,   256, 0, stream>>>(W,   Wb,   H_DIM * 256 / 4);
    k_cvt_bf16<<<(H_DIM * H_DIM / 4 + 255) / 256, 256, 0, stream>>>(Wf1, Wf1b, H_DIM * H_DIM / 4);
    k_cvt_bf16<<<(64 * H_DIM / 4 + 255) / 256,    256, 0, stream>>>(Wf2, Wf2b, 64 * H_DIM / 4);

    // 2) row.min()
    k_row_min<<<512, 256, 0, stream>>>(edge, E, rmin);

    // 3) transpose WA -> [N, 128] so per-edge gathers/scatters are contiguous
    k_transpose_WA<<<dim3((N + 31) / 32, H_DIM / 32), dim3(32, 8), 0, stream>>>(WA, WAt, N);

    // 4) sparse aggregation: xA = segment_sum(WA.T[col], row) + bA
    long long tot = (long long)E * 32;
    k_scatter_xA<<<(unsigned)((tot + 255) / 256), 256, 0, stream>>>(edge, E, WAt, xAX, rmin);

    const unsigned gx = (unsigned)((N / 16 + 3) / 4);   // 4 row-strips / block

    // 5) xX = x @ WX^T + bX -> columns 128..255 of xAX (free concat)
    k_wmma_gemm<8, 0, false><<<gx, 128, 0, stream>>>(
        x, IN, WXb, IN, bX, nullptr, 0, nullptr, nullptr, xAX, 256, H_DIM, N);

    // 6) h = relu([xA|xX] @ W^T + bW + xA + xX)
    k_wmma_gemm<8, 2, false><<<gx, 128, 0, stream>>>(
        xAX, 256, Wb, 256, bW, xAX, 256, nullptr, nullptr, hbuf, H_DIM, 0, N);

    // 7) h2 = relu(h @ Wf1^T + bf1)
    k_wmma_gemm<8, 1, false><<<gx, 128, 0, stream>>>(
        hbuf, H_DIM, Wf1b, H_DIM, bf1, nullptr, 0, nullptr, nullptr, h2, H_DIM, 0, N);

    // 8) BN stats + coefficients
    k_bn_stats<<<200, 128, 0, stream>>>(h2, sum, sumsq, N);
    k_bn_coeffs<<<1, 128, 0, stream>>>(sum, sumsq, gamma, beta, scale, shift, N);

    // 9) out = BN(h2) @ Wf2^T + bf2   (BN fused into A-load path)
    k_wmma_gemm<4, 0, true><<<gx, 128, 0, stream>>>(
        h2, H_DIM, Wf2b, H_DIM, bf2, nullptr, 0, scale, shift, out, 64, 0, N);
}
